// Head_11046655885791
// MI455X (gfx1250) — matrix operand order repaired and checked by reading, not compiled, not run on hardware
//
#include <hip/hip_runtime.h>
#include <hip/hip_bf16.h>

typedef __attribute__((ext_vector_type(16))) _Float16 v16h;
typedef __attribute__((ext_vector_type(8)))  _Float16 v8h;
typedef __attribute__((ext_vector_type(8)))  float    v8f;
typedef __attribute__((ext_vector_type(4)))  float    f4;

#define B_SZ 8
#define T_SZ 4096
#define D_SZ 128
#define H_SZ 64

static __device__ __forceinline__ v16h cat8(v8h lo, v8h hi) {
    return __builtin_shufflevector(lo, hi, 0, 1, 2, 3, 4, 5, 6, 7,
                                           8, 9, 10, 11, 12, 13, 14, 15);
}

// ---------------------------------------------------------------------------
// Kernel 1: QKV projection.  x:[B*T,128] f32 @ W:[128,64] f32 -> f16 outputs.
// Q,K row-major [B*T,64]; V transposed [B][64][T].
// Blocks 0..511 -> Wq, 512..1023 -> Wk, 1024..1535 -> Wv. 4 waves/block,
// each wave does one 16-row tile (all 4 column tiles, reusing A fragments).
// ---------------------------------------------------------------------------
__global__ __launch_bounds__(128) void qkv_proj_kernel(
    const float* __restrict__ x,
    const float* __restrict__ Wq, const float* __restrict__ Wk,
    const float* __restrict__ Wv,
    _Float16* __restrict__ Qh, _Float16* __restrict__ Kh,
    _Float16* __restrict__ Vt)
{
    __shared__ _Float16 wlds[64 * 136];   // W^T, padded row stride 136 halves

    const int tid  = threadIdx.x;
    const int lane = tid & 31;
    const int wid  = tid >> 5;
    const int mat  = blockIdx.x >> 9;              // 0=Q 1=K 2=V (uniform per block)
    const int rtb  = (blockIdx.x & 511) * 4;       // row-tile base for block

    const float* W = (mat == 0) ? Wq : ((mat == 1) ? Wk : Wv);

    // Cooperative load of W (128x64 f32) into LDS transposed as f16.
    for (int idx = tid; idx < 128 * 64; idx += 128) {
        int k = idx >> 6;
        int n = idx & 63;
        wlds[n * 136 + k] = (_Float16)W[idx];
    }
    __syncthreads();

    const int rowTile = rtb + wid;                 // 0..2047
    const int row0    = rowTile * 16;
    const int nloc    = lane & 15;                 // m for A, n for B/C
    const int g       = lane >> 4;

    // A fragments: x rows row0..row0+15, K = 128 -> 4 k-steps of 32.
    // Per lane: halves 0..7 <-> k = base+g*8+(0..7); halves 8..15 <-> +16.
    v16h a[4];
    const float* xrow = x + (size_t)(row0 + nloc) * D_SZ;
#pragma unroll
    for (int ks = 0; ks < 4; ++ks) {
        f4 c0 = *(const f4*)(xrow + ks * 32 + g * 8);
        f4 c1 = *(const f4*)(xrow + ks * 32 + g * 8 + 4);
        f4 c2 = *(const f4*)(xrow + ks * 32 + 16 + g * 8);
        f4 c3 = *(const f4*)(xrow + ks * 32 + 16 + g * 8 + 4);
#pragma unroll
        for (int i = 0; i < 4; ++i) {
            a[ks][i]      = (_Float16)c0[i];
            a[ks][4 + i]  = (_Float16)c1[i];
            a[ks][8 + i]  = (_Float16)c2[i];
            a[ks][12 + i] = (_Float16)c3[i];
        }
    }

#pragma unroll
    for (int nt = 0; nt < 4; ++nt) {
        v8f c = {};
#pragma unroll
        for (int ks = 0; ks < 4; ++ks) {
            // B fragment: halves 0..7 <-> k = ks*32 + g*8 + (0..7); halves 8..15 <-> +16
            const _Float16* wp = wlds + (nt * 16 + nloc) * 136 + ks * 32 + g * 8;
            v16h bf = cat8(*(const v8h*)(wp), *(const v8h*)(wp + 16));
            c = __builtin_amdgcn_wmma_f32_16x16x32_f16(
                    false, a[ks], false, bf, (short)0, c, false, false);
        }
        if (mat < 2) {
            _Float16* outp = (mat == 0) ? Qh : Kh;
#pragma unroll
            for (int r = 0; r < 8; ++r) {
                int mm = r + 8 * g;
                outp[(size_t)(row0 + mm) * H_SZ + nt * 16 + nloc] = (_Float16)c[r];
            }
        } else {
            // V^T: lane's 8 rows map to 8 consecutive t -> one b128 store.
            v8h pk;
#pragma unroll
            for (int r = 0; r < 8; ++r) pk[r] = (_Float16)c[r];
            int grow = row0 + 8 * g;               // first of 8 consecutive rows
            int b    = grow >> 12;
            int t    = grow & (T_SZ - 1);
            *(v8h*)(Vt + ((size_t)b * H_SZ + nt * 16 + nloc) * T_SZ + t) = pk;
        }
    }
}

// ---------------------------------------------------------------------------
// Kernel 2: fused causal attention with online softmax (flash-attention).
// 512 blocks (B * T/64), 4 waves/block, each wave owns a 16-query tile.
// ---------------------------------------------------------------------------
__global__ __launch_bounds__(128) void flash_attn_kernel(
    const _Float16* __restrict__ Qh, const _Float16* __restrict__ Kh,
    const _Float16* __restrict__ Vt, float* __restrict__ out)
{
    __shared__ _Float16 pbuf[4][16 * 40];          // per-wave P tile, padded

    const int tid  = threadIdx.x;
    const int lane = tid & 31;
    const int wid  = tid >> 5;
    const int b    = blockIdx.x >> 6;              // batch
    const int qt   = blockIdx.x & 63;
    const int q0   = qt * 64 + wid * 16;           // first query row of wave
    const int n    = lane & 15;
    const int g    = lane >> 4;
    const float scale = 0.08838834764831845f;      // 1/sqrt(128)  (ref uses sqrt(D))

    // Q A-fragments (K-dim 64 -> 2 halves), each = two aligned b128 loads.
    v16h qa[2];
    const _Float16* qrow = Qh + ((size_t)b * T_SZ + q0 + n) * H_SZ;
#pragma unroll
    for (int kh = 0; kh < 2; ++kh) {
        const _Float16* qp = qrow + kh * 32;
        qa[kh] = cat8(*(const v8h*)(qp + g * 8),
                      *(const v8h*)(qp + 16 + g * 8));
    }

    v8f o[4] = {};
    float mrun[8], lrun[8];
#pragma unroll
    for (int r = 0; r < 8; ++r) { mrun[r] = -__builtin_inff(); lrun[r] = 0.0f; }

    _Float16* pb = &pbuf[wid][0];

    for (int kt = 0; kt < q0 + 16; kt += 32) {
        // ---- S = Q @ K^T  (two 16-key column tiles) ----
        v8f s[2];
#pragma unroll
        for (int nt2 = 0; nt2 < 2; ++nt2) {
            v8f c = {};
            const _Float16* krow = Kh + ((size_t)b * T_SZ + kt + nt2 * 16 + n) * H_SZ;
#pragma unroll
            for (int kh = 0; kh < 2; ++kh) {
                const _Float16* kp = krow + kh * 32 + g * 8;
                v16h kb = cat8(*(const v8h*)(kp), *(const v8h*)(kp + 16));
                c = __builtin_amdgcn_wmma_f32_16x16x32_f16(
                        false, qa[kh], false, kb, (short)0, c, false, false);
            }
            s[nt2] = c;
        }

        // ---- scale + causal mask ----
        const bool needMask = (kt + 31) > q0;      // uniform per wave
#pragma unroll
        for (int nt2 = 0; nt2 < 2; ++nt2) {
#pragma unroll
            for (int r = 0; r < 8; ++r) {
                int qq = q0 + r + 8 * g;
                int kk = kt + nt2 * 16 + n;
                float v = s[nt2][r] * scale;
                if (needMask && kk > qq) v = -__builtin_inff();
                s[nt2][r] = v;
            }
        }

        // ---- online softmax stats (row lives in one 16-lane half) ----
        float mnew[8], alpha[8];
#pragma unroll
        for (int r = 0; r < 8; ++r) {
            float t = fmaxf(s[0][r], s[1][r]);
            t = fmaxf(t, __shfl_xor(t, 8, 32));
            t = fmaxf(t, __shfl_xor(t, 4, 32));
            t = fmaxf(t, __shfl_xor(t, 2, 32));
            t = fmaxf(t, __shfl_xor(t, 1, 32));
            mnew[r]  = fmaxf(mrun[r], t);
            alpha[r] = __expf(mrun[r] - mnew[r]);
            mrun[r]  = mnew[r];
        }

        // ---- P = exp(S - m), row sums, stage P (f16) in LDS ----
#pragma unroll
        for (int r = 0; r < 8; ++r) {
            float p0 = __expf(s[0][r] - mnew[r]);
            float p1 = __expf(s[1][r] - mnew[r]);
            int mm = r + 8 * g;
            pb[mm * 40 + n]      = (_Float16)p0;
            pb[mm * 40 + 16 + n] = (_Float16)p1;
            float ps = p0 + p1;
            ps += __shfl_xor(ps, 8, 32);
            ps += __shfl_xor(ps, 4, 32);
            ps += __shfl_xor(ps, 2, 32);
            ps += __shfl_xor(ps, 1, 32);
            lrun[r] = lrun[r] * alpha[r] + ps;
        }

        // ---- rescale running O ----
#pragma unroll
        for (int t4 = 0; t4 < 4; ++t4)
#pragma unroll
            for (int r = 0; r < 8; ++r)
                o[t4][r] *= alpha[r];

        // ---- reload P in A-fragment layout (wave-local LDS round trip) ----
        const _Float16* pp = pb + n * 40;
        v16h pa = cat8(*(const v8h*)(pp + g * 8),
                       *(const v8h*)(pp + 16 + g * 8));

        // ---- O += P @ V  (V stored transposed -> contiguous B fragments) ----
#pragma unroll
        for (int t4 = 0; t4 < 4; ++t4) {
            const _Float16* vp =
                Vt + ((size_t)b * H_SZ + t4 * 16 + n) * T_SZ + kt + g * 8;
            v16h vb = cat8(*(const v8h*)(vp), *(const v8h*)(vp + 16));
            o[t4] = __builtin_amdgcn_wmma_f32_16x16x32_f16(
                        false, pa, false, vb, (short)0, o[t4], false, false);
        }
    }

    // ---- finalize: multiply by 1/rowsum (fast rcp), write fp32 ----
    float rinv[8];
#pragma unroll
    for (int r = 0; r < 8; ++r) rinv[r] = __builtin_amdgcn_rcpf(lrun[r]);
#pragma unroll
    for (int t4 = 0; t4 < 4; ++t4) {
#pragma unroll
        for (int r = 0; r < 8; ++r) {
            int mm = r + 8 * g;
            out[((size_t)b * T_SZ + q0 + mm) * H_SZ + t4 * 16 + n] =
                o[t4][r] * rinv[r];
        }
    }
}

// ---------------------------------------------------------------------------
extern "C" void kernel_launch(void* const* d_in, const int* in_sizes, int n_in,
                              void* d_out, int out_size, void* d_ws, size_t ws_size,
                              hipStream_t stream) {
    const float* x  = (const float*)d_in[0];
    const float* Wq = (const float*)d_in[1];
    const float* Wk = (const float*)d_in[2];
    const float* Wv = (const float*)d_in[3];
    float* out = (float*)d_out;

    // Workspace: Q, K (row-major f16), V^T (f16): 3 * B*T*64 halves = 12 MB.
    const size_t elems = (size_t)B_SZ * T_SZ * H_SZ;
    _Float16* Qh = (_Float16*)d_ws;
    _Float16* Kh = Qh + elems;
    _Float16* Vt = Kh + elems;

    qkv_proj_kernel<<<1536, 128, 0, stream>>>(x, Wq, Wk, Wv, Qh, Kh, Vt);
    flash_attn_kernel<<<512, 128, 0, stream>>>(Qh, Kh, Vt, out);
}
